// NNP_47236050321984
// MI455X (gfx1250) — matrix operand order repaired and checked by reading, not compiled, not run on hardware
//
#include <hip/hip_runtime.h>
#include <hip/hip_bf16.h>

typedef __attribute__((ext_vector_type(16))) _Float16 v16h;
typedef __attribute__((ext_vector_type(8)))  _Float16 v8h;
typedef __attribute__((ext_vector_type(8)))  float    v8f;
typedef __attribute__((ext_vector_type(4)))  float    v4f;

#define NOS 4
#define INF 128   // descriptor size
#define H1  32
#define H2  32

static __device__ __forceinline__ float dev_tanh(float x) {
#if __has_builtin(__builtin_amdgcn_tanhf)
  return __builtin_amdgcn_tanhf(x);     // v_tanh_f32 on gfx1250
#else
  return tanhf(x);
#endif
}

// Load 16 f16 as two 16B chunks (alignment-safe; lowers to ds_load_b128 x2)
static __device__ __forceinline__ v16h ld_v16h(const _Float16* p) {
  v8h a = *(const v8h*)p;
  v8h b = *(const v8h*)(p + 8);
  v16h r;
#pragma unroll
  for (int i = 0; i < 8; ++i) { r[i] = a[i]; r[i + 8] = b[i]; }
  return r;
}

__global__ __launch_bounds__(256)
void nnp_moe_wmma(const float* __restrict__ desc,
                  const int*   __restrict__ species,
                  const float* __restrict__ W0,
                  const float* __restrict__ b0,
                  const float* __restrict__ W1,
                  const float* __restrict__ b1,
                  const float* __restrict__ W2,
                  const float* __restrict__ b2,
                  float* __restrict__ out,
                  int n_atoms, int ntiles) {
  // LDS staging: weights transposed to f16 [species][n][k] so WMMA B fragments
  // are contiguous per lane (B order used here: lane = column n, 16 consecutive
  // K halves; upper half-wave starts at K+16).
  __shared__ _Float16 W0T[NOS * H1 * INF];   // 32 KB
  __shared__ _Float16 W1T[NOS * H2 * H1];    //  8 KB
  __shared__ float    b0L[NOS * H1];
  __shared__ float    b1L[NOS * H2];
  __shared__ float    w2L[NOS * H2];
  __shared__ float    b2L[NOS];
  __shared__ _Float16 h0buf[8 * 16 * H1];    // per-wave 16x32 relayout tile, 8 KB
  __shared__ float    red[8];

  // ---- cooperative weight staging (f32 global -> f16 LDS, transposed) ----
  for (int idx = threadIdx.x; idx < NOS * INF * H1; idx += 256) {
    int s = idx >> 12;            // 128*32 = 4096 per species
    int rem = idx & 4095;
    int k = rem >> 5;             // 0..127
    int n = rem & 31;
    W0T[(s * H1 + n) * INF + k] = (_Float16)W0[idx];
  }
  for (int idx = threadIdx.x; idx < NOS * H1 * H2; idx += 256) {
    int s = idx >> 10;            // 32*32 = 1024 per species
    int rem = idx & 1023;
    int k = rem >> 5;
    int n = rem & 31;
    W1T[(s * H2 + n) * H1 + k] = (_Float16)W1[idx];
  }
  for (int idx = threadIdx.x; idx < NOS * H1; idx += 256) {
    b0L[idx] = b0[idx];
    b1L[idx] = b1[idx];
    w2L[idx] = W2[idx];           // W2 is [NOS][H2][1] -> flat [NOS*H2]
  }
  if (threadIdx.x < NOS) b2L[threadIdx.x] = b2[threadIdx.x];
  __syncthreads();

  const int wave  = threadIdx.x >> 5;
  const int lane  = threadIdx.x & 31;
  const int nlo   = lane & 15;
  const bool hig  = lane >= 16;
  const int mrow  = hig ? 8 : 0;     // C-layout row base for this half-wave
  const int koff8 = hig ? 16 : 0;    // A-layout K base for this half-wave
  _Float16* hb = h0buf + wave * 16 * H1;

  float acc = 0.0f;

  // Keep one tile in flight per wave: prevents the cross-tile software
  // pipelining that pushed allocation past 256 VGPRs (s_set_vgpr_msb noise,
  // ~3 waves/SIMD). Bandwidth-bound kernel -> occupancy wins.
#pragma unroll 1
  for (int tile = blockIdx.x * 8 + wave; tile < ntiles; tile += gridDim.x * 8) {
    const int base = tile * 16;
    int arow = base + nlo;
    bool valid = arow < n_atoms;
    if (!valid) arow = n_atoms - 1;            // clamp (contribution masked)

    // prefetch next grid-stride tile's row
    {
      int nt = tile + gridDim.x * 8;
      if (nt < ntiles)
        __builtin_prefetch(desc + (size_t)(nt * 16 + nlo) * INF, 0, 0);
    }

    // ---- A fragments: 4 K-chunks of 32, f32 -> f16 in registers ----
    // desc is a 256MB single-use stream (> 192MB L2): non-temporal loads keep
    // it from evicting the hot weights / other tiles.
    const float* rowp = desc + (size_t)arow * INF + koff8;
    v16h A[4];
#pragma unroll
    for (int kc = 0; kc < 4; ++kc) {
      v4f f0 = __builtin_nontemporal_load((const v4f*)(rowp + kc * 32));
      v4f f1 = __builtin_nontemporal_load((const v4f*)(rowp + kc * 32 + 4));
      v4f f2 = __builtin_nontemporal_load((const v4f*)(rowp + kc * 32 + 8));
      v4f f3 = __builtin_nontemporal_load((const v4f*)(rowp + kc * 32 + 12));
      v16h a;
#pragma unroll
      for (int i = 0; i < 4; ++i) {
        a[i]      = (_Float16)f0[i];
        a[4 + i]  = (_Float16)f1[i];
        a[8 + i]  = (_Float16)f2[i];
        a[12 + i] = (_Float16)f3[i];
      }
      A[kc] = a;
    }

    const int sp = valid ? species[arow] : -1;

    // ---- all 4 expert MLPs on this 16-atom tile, mask by species ----
    // unroll 1: one species' fragments live at a time (~150 VGPRs total).
#pragma unroll 1
    for (int s = 0; s < NOS; ++s) {
      const _Float16* w0s = W0T + s * H1 * INF;

      // Layer 0: [16x128] x [128x32] -> two 16x16 accumulators
      v8f c0 = {}, c1 = {};
#pragma unroll
      for (int kc = 0; kc < 4; ++kc) {
        const int kb = kc * 32 + (hig ? 16 : 0);
        v16h B0 = ld_v16h(w0s + nlo * INF + kb);
        v16h B1 = ld_v16h(w0s + (16 + nlo) * INF + kb);
        c0 = __builtin_amdgcn_wmma_f32_16x16x32_f16(false, A[kc], false, B0,
                                                    (short)0, c0, false, false);
        c1 = __builtin_amdgcn_wmma_f32_16x16x32_f16(false, A[kc], false, B1,
                                                    (short)0, c1, false, false);
      }
      const float bn0 = b0L[s * H1 + nlo];
      const float bn1 = b0L[s * H1 + 16 + nlo];
#pragma unroll
      for (int r = 0; r < 8; ++r) {
        c0[r] = dev_tanh(c0[r] + bn0);
        c1[r] = dev_tanh(c1[r] + bn1);
      }

      // Relayout h0 (C layout -> A layout) through per-wave LDS tile.
#pragma unroll
      for (int r = 0; r < 8; ++r) {
        hb[(r + mrow) * H1 + nlo]      = (_Float16)c0[r];
        hb[(r + mrow) * H1 + 16 + nlo] = (_Float16)c1[r];
      }
      v8h lo8 = *(const v8h*)(hb + nlo * H1 + koff8);
      v8h hi8 = *(const v8h*)(hb + nlo * H1 + koff8 + 8);
      v16h A1;
#pragma unroll
      for (int i = 0; i < 8; ++i) { A1[i] = lo8[i]; A1[8 + i] = hi8[i]; }

      // Layer 1: [16x32] x [32x32]
      const _Float16* w1s = W1T + s * H2 * H1;
      const int kb1 = hig ? 16 : 0;
      v16h B10 = ld_v16h(w1s + nlo * H1 + kb1);
      v16h B11 = ld_v16h(w1s + (16 + nlo) * H1 + kb1);
      v8f z = {};
      v8f d0 = __builtin_amdgcn_wmma_f32_16x16x32_f16(false, A1, false, B10,
                                                      (short)0, z, false, false);
      v8f d1 = __builtin_amdgcn_wmma_f32_16x16x32_f16(false, A1, false, B11,
                                                      (short)0, z, false, false);

      const float cb0 = b1L[s * H2 + nlo];
      const float cb1 = b1L[s * H2 + 16 + nlo];
      const float w2a = w2L[s * H2 + nlo];
      const float w2b = w2L[s * H2 + 16 + nlo];
      const float bb2 = b2L[s];

      // Layer 2 + species mask. Final answer is a global sum, so each lane
      // just accumulates its partial dot; b2 added once per selected atom.
#pragma unroll
      for (int r = 0; r < 8; ++r) {
        float h1a = dev_tanh(d0[r] + cb0);
        float h1b = dev_tanh(d1[r] + cb1);
        float e = h1a * w2a + h1b * w2b;
        int spm = __shfl(sp, r + mrow, 32);
        float sel = (spm == s) ? 1.0f : 0.0f;
        acc += sel * e;
        if (nlo == 0 && spm == s) acc += bb2;
      }
    }
  }

  // ---- reduction: wave shfl-xor -> LDS -> one atomic per block ----
#pragma unroll
  for (int off = 16; off > 0; off >>= 1)
    acc += __shfl_xor(acc, off, 32);
  if (lane == 0) red[wave] = acc;
  __syncthreads();
  if (threadIdx.x == 0) {
    float t = 0.0f;
#pragma unroll
    for (int w = 0; w < 8; ++w) t += red[w];
    atomicAdd(out, t);
  }
}

extern "C" void kernel_launch(void* const* d_in, const int* in_sizes, int n_in,
                              void* d_out, int out_size, void* d_ws, size_t ws_size,
                              hipStream_t stream) {
  const float* desc    = (const float*)d_in[0];
  const int*   species = (const int*)  d_in[1];
  const float* W0      = (const float*)d_in[2];
  const float* b0      = (const float*)d_in[3];
  const float* W1      = (const float*)d_in[4];
  const float* b1      = (const float*)d_in[5];
  const float* W2      = (const float*)d_in[6];
  const float* b2      = (const float*)d_in[7];
  float* out = (float*)d_out;

  const int n_atoms = in_sizes[0] / INF;
  const int ntiles  = (n_atoms + 15) / 16;

  hipMemsetAsync(out, 0, sizeof(float), stream);

  int blocks = (ntiles + 7) / 8;
  if (blocks > 1024) blocks = 1024;
  if (blocks < 1) blocks = 1;

  nnp_moe_wmma<<<blocks, 256, 0, stream>>>(desc, species, W0, b0, W1, b1, W2, b2,
                                           out, n_atoms, ntiles);
}
